// GIN_73753178406913
// MI455X (gfx1250) — matrix operand order repaired and checked by reading, not compiled, not run on hardware
//
#include <hip/hip_runtime.h>
#include <hip/hip_bf16.h>

typedef __attribute__((ext_vector_type(16))) _Float16 v16h;
typedef __attribute__((ext_vector_type(8)))  float    v8f;

#define GIN_N_NODES 100000
#define GIN_N_EDGES 1200000
#define GIN_HIDDEN  64
#define GIN_LATENT  32
#define GIN_NGRAPHS 512
#define GIN_BN_EPS  1e-5f
#define GIN_K       64          // all GEMMs have K = 64
#define GIN_KSTEPS  2           // K / 32
#define GIN_WSLOT   4096        // packed halfs reserved per weight matrix

// ---------------------------------------------------------------------------
// Pack W[K x Nout] (f32 row-major) into the CDNA5 WMMA B-fragment layout
// (ISA 7.12.2, wave32, 16-bit B 32x16): lane(0..15) = col N, holds K={0..7,16..23} in
// VGPR pairs; lane(16..31) = col N, holds K={8..15,24..31}. Packed order:
//   Wp[(((nt*KSTEPS)+ks)*32 + lane)*16 + i]  (f16)
// so each lane reads one contiguous 32-byte chunk per k-step.
// ---------------------------------------------------------------------------
__global__ void pack_w(const float* __restrict__ W, _Float16* __restrict__ Wp, int Nout)
{
    const int t = blockIdx.x * blockDim.x + threadIdx.x;
    const int total = (Nout / 16) * GIN_KSTEPS * 32 * 16;
    if (t >= total) return;
    const int i    = t & 15;
    const int lane = (t >> 4) & 31;
    const int rem  = t >> 9;
    const int ks   = rem % GIN_KSTEPS;
    const int nt   = rem / GIN_KSTEPS;
    const int lane_lo = lane & 15;
    const int lane_hi = lane >> 4;
    const int j = i >> 1, p = i & 1;
    const int k = ks * 32 + (i < 8 ? 8 * lane_hi + i : 16 + 8 * lane_hi + (i - 8));
    const int n = nt * 16 + lane_lo;
    Wp[t] = (_Float16)W[(size_t)k * Nout + n];
}

// ---------------------------------------------------------------------------
// Y[M,Nout] = epilogue( X[M,64] @ W + b ); W comes pre-packed (see pack_w).
// One wave = one 16x16 output tile; blockDim.x/32 waves tile Nout; blockIdx.x
// tiles M in strips of 16. REQUIRES M % 16 == 0 (true for 100000 and 512), so
// EXEC stays all-1s around the WMMAs and no load guards are needed.
// A-fragment (16x32 f16): lane(0..15) row M=lane, K groups {0..7,16..23};
// lane(16..31) row M=lane-16, K groups {8..15,24..31} -> two contiguous runs
// of 8 floats per k-step -> four float4 loads.
// ---------------------------------------------------------------------------
__global__ void lin_wmma(const float* __restrict__ X,
                         const _Float16* __restrict__ Wp,
                         const float* __restrict__ bias,
                         const float* __restrict__ prelu_a, int use_prelu,
                         const float* __restrict__ bn_g, const float* __restrict__ bn_b,
                         const float* __restrict__ bn_m, const float* __restrict__ bn_v,
                         int use_bn,
                         float* __restrict__ Y, int Nout)
{
    const int wave    = threadIdx.x >> 5;
    const int lane    = threadIdx.x & 31;
    const int lane_lo = lane & 15;
    const int lane_hi = lane >> 4;
    const int row0    = blockIdx.x * 16;
    const int n       = wave * 16 + lane_lo;        // B/D column for this lane

    const float*    xrow = X + (size_t)(row0 + lane_lo) * GIN_K;
    const _Float16* wp   = Wp + ((size_t)(wave * GIN_KSTEPS) * 32 + lane) * 16;

    v8f acc = {};
#pragma unroll
    for (int ks = 0; ks < GIN_KSTEPS; ++ks) {
        const int kb0 = ks * 32 + (lane_hi ? 8 : 0);  // first run of 8 K values
        const int kb1 = kb0 + 16;                     // second run of 8 K values
        const float4 a0 = *(const float4*)(xrow + kb0);
        const float4 a1 = *(const float4*)(xrow + kb0 + 4);
        const float4 a2 = *(const float4*)(xrow + kb1);
        const float4 a3 = *(const float4*)(xrow + kb1 + 4);
        v16h afrag;
        afrag[0]  = (_Float16)a0.x; afrag[1]  = (_Float16)a0.y;
        afrag[2]  = (_Float16)a0.z; afrag[3]  = (_Float16)a0.w;
        afrag[4]  = (_Float16)a1.x; afrag[5]  = (_Float16)a1.y;
        afrag[6]  = (_Float16)a1.z; afrag[7]  = (_Float16)a1.w;
        afrag[8]  = (_Float16)a2.x; afrag[9]  = (_Float16)a2.y;
        afrag[10] = (_Float16)a2.z; afrag[11] = (_Float16)a2.w;
        afrag[12] = (_Float16)a3.x; afrag[13] = (_Float16)a3.y;
        afrag[14] = (_Float16)a3.z; afrag[15] = (_Float16)a3.w;

        const v16h bfrag = *(const v16h*)(wp + (size_t)ks * 32 * 16);

        // (neg_a, A, neg_b, B, c_mod, C, reuse_a, reuse_b)
        acc = __builtin_amdgcn_wmma_f32_16x16x32_f16(false, afrag, false, bfrag,
                                                     (short)0, acc, false, false);
    }

    const float bcol = bias[n];
    const float aP   = use_prelu ? prelu_a[0] : 1.0f;
    float sc = 1.0f, mu = 0.0f, bt = 0.0f;
    if (use_bn) {
        sc = bn_g[n] * rsqrtf(bn_v[n] + GIN_BN_EPS);
        mu = bn_m[n];
        bt = bn_b[n];
    }
#pragma unroll
    for (int v = 0; v < 8; ++v) {
        const int row = row0 + v + lane_hi * 8;       // D layout: M = v + 8*(lane>=16)
        float t = acc[v] + bcol;
        if (use_prelu) t = (t >= 0.0f) ? t : aP * t;
        if (use_bn)    t = (t - mu) * sc + bt;
        Y[(size_t)row * Nout + n] = t;
    }
}

__global__ void copy_f4(const float4* __restrict__ src, float4* __restrict__ dst, int n) {
    const int i = blockIdx.x * blockDim.x + threadIdx.x;
    if (i < n) dst[i] = src[i];
}

__global__ void zero_f(float* __restrict__ p, int n) {
    const int i = blockIdx.x * blockDim.x + threadIdx.x;
    if (i < n) p[i] = 0.0f;
}

// z[dst] += h[src] over edges; 16 lanes per edge, float4 gather + 4 f32 atomics.
__global__ void edge_scatter(const float* __restrict__ h,
                             const long long* __restrict__ src,
                             const long long* __restrict__ dst,
                             float* __restrict__ z, int n_edges)
{
    const int t  = blockIdx.x * blockDim.x + threadIdx.x;
    const int e  = t >> 4;
    const int c4 = (t & 15) * 4;
    if (e >= n_edges) return;
    const long long s = src[e];
    const long long d = dst[e];
    const float4 v = *(const float4*)(h + (size_t)s * GIN_HIDDEN + c4);
    float* zp = z + (size_t)d * GIN_HIDDEN + c4;
    atomicAdd(zp + 0, v.x);
    atomicAdd(zp + 1, v.y);
    atomicAdd(zp + 2, v.z);
    atomicAdd(zp + 3, v.w);
}

// pool[batch[node]] += h[node]
__global__ void pool_scatter(const float* __restrict__ h,
                             const long long* __restrict__ batch,
                             float* __restrict__ pool, int n_nodes)
{
    const int t    = blockIdx.x * blockDim.x + threadIdx.x;
    const int node = t >> 4;
    const int c4   = (t & 15) * 4;
    if (node >= n_nodes) return;
    const long long g = batch[node];
    const float4 v = *(const float4*)(h + (size_t)node * GIN_HIDDEN + c4);
    float* pp = pool + (size_t)g * GIN_HIDDEN + c4;
    atomicAdd(pp + 0, v.x);
    atomicAdd(pp + 1, v.y);
    atomicAdd(pp + 2, v.z);
    atomicAdd(pp + 3, v.w);
}

__global__ void bn_inplace(float* __restrict__ p,
                           const float* __restrict__ g, const float* __restrict__ b,
                           const float* __restrict__ m, const float* __restrict__ v,
                           int total, int dim)
{
    const int i = blockIdx.x * blockDim.x + threadIdx.x;
    if (i >= total) return;
    const int c = i % dim;
    const float sc = g[c] * rsqrtf(v[c] + GIN_BN_EPS);
    p[i] = (p[i] - m[c]) * sc + b[c];
}

extern "C" void kernel_launch(void* const* d_in, const int* in_sizes, int n_in,
                              void* d_out, int out_size, void* d_ws, size_t ws_size,
                              hipStream_t stream)
{
    (void)in_sizes; (void)n_in; (void)out_size; (void)ws_size;

    // Flattened input order (insertion order of setup_inputs dicts):
    // 0:x 1:edge_index(i64) 2:batch(i64) 3:fc0.W 4:fc0.b
    // per layer l (base=5+14*l): lin1.W lin1.b a1 bn_in.{g,b,m,v} lin2.W lin2.b a2 bn_out.{g,b,m,v}
    // 47..50: bn_final.{g,b,m,v}  51:fc.W  52:fc.b
    const float*     x     = (const float*)d_in[0];
    const long long* edge  = (const long long*)d_in[1];
    const long long* batch = (const long long*)d_in[2];
    const float*     fc0_W = (const float*)d_in[3];
    const float*     fc0_b = (const float*)d_in[4];
    auto Lp = [&](int l, int off) { return (const float*)d_in[5 + 14 * l + off]; };
    const float* bnf_g = (const float*)d_in[47];
    const float* bnf_b = (const float*)d_in[48];
    const float* bnf_m = (const float*)d_in[49];
    const float* bnf_v = (const float*)d_in[50];
    const float* fc_W  = (const float*)d_in[51];
    const float* fc_b  = (const float*)d_in[52];

    const long long* esrc = edge;               // edge_index[0]
    const long long* edst = edge + GIN_N_EDGES; // edge_index[1]

    float* buf0 = (float*)d_ws;                               // h
    float* buf1 = buf0 + (size_t)GIN_N_NODES * GIN_HIDDEN;    // z
    float* pool = buf1 + (size_t)GIN_N_NODES * GIN_HIDDEN;
    _Float16* wpk = (_Float16*)(pool + (size_t)GIN_NGRAPHS * GIN_HIDDEN);
    // packed weight slots: 0=fc0, 1+2l=lin1(l), 2+2l=lin2(l), 7=fc
    auto Wslot = [&](int s) { return wpk + (size_t)s * GIN_WSLOT; };

    // --- pack all 8 weight matrices into B-fragment layout (tiny, one-shot) ---
    {
        const int tot64 = (GIN_HIDDEN / 16) * GIN_KSTEPS * 512;  // 4096
        const int tot32 = (GIN_LATENT / 16) * GIN_KSTEPS * 512;  // 2048
        pack_w<<<(tot64 + 255) / 256, 256, 0, stream>>>(fc0_W, Wslot(0), GIN_HIDDEN);
        for (int l = 0; l < 3; ++l) {
            pack_w<<<(tot64 + 255) / 256, 256, 0, stream>>>(Lp(l, 0), Wslot(1 + 2 * l),
                                                            GIN_HIDDEN);
            pack_w<<<(tot64 + 255) / 256, 256, 0, stream>>>(Lp(l, 7), Wslot(2 + 2 * l),
                                                            GIN_HIDDEN);
        }
        pack_w<<<(tot32 + 255) / 256, 256, 0, stream>>>(fc_W, Wslot(7), GIN_LATENT);
    }

    const int linGrid = GIN_N_NODES / 16;   // 6250 strips, exact multiple of 16

    // h = x @ fc0.W + fc0.b
    lin_wmma<<<linGrid, 128, 0, stream>>>(x, Wslot(0), fc0_b,
                                          nullptr, 0,
                                          nullptr, nullptr, nullptr, nullptr, 0,
                                          buf0, GIN_HIDDEN);

    for (int l = 0; l < 3; ++l) {
        // z = h (copy), then z[dst] += h[src]  =>  z = h + agg
        const int ncopy4 = GIN_N_NODES * GIN_HIDDEN / 4;
        copy_f4<<<(ncopy4 + 255) / 256, 256, 0, stream>>>((const float4*)buf0,
                                                          (float4*)buf1, ncopy4);
        const int tScatter = GIN_N_EDGES * 16;
        edge_scatter<<<(tScatter + 255) / 256, 256, 0, stream>>>(buf0, esrc, edst,
                                                                 buf1, GIN_N_EDGES);
        // h' = bn_in(prelu(z @ lin1.W + b, a1))
        lin_wmma<<<linGrid, 128, 0, stream>>>(buf1, Wslot(1 + 2 * l), Lp(l, 1),
                                              Lp(l, 2), 1,
                                              Lp(l, 3), Lp(l, 4), Lp(l, 5), Lp(l, 6), 1,
                                              buf0, GIN_HIDDEN);
        // h = bn_out(prelu(h' @ lin2.W + b, a2))
        lin_wmma<<<linGrid, 128, 0, stream>>>(buf0, Wslot(2 + 2 * l), Lp(l, 8),
                                              Lp(l, 9), 1,
                                              Lp(l, 10), Lp(l, 11), Lp(l, 12), Lp(l, 13), 1,
                                              buf1, GIN_HIDDEN);
        float* tmp = buf0; buf0 = buf1; buf1 = tmp;  // h lives in buf0 again
    }

    // global_add_pool + bn_final + fc
    const int npool = GIN_NGRAPHS * GIN_HIDDEN;
    zero_f<<<(npool + 255) / 256, 256, 0, stream>>>(pool, npool);
    const int tPool = GIN_N_NODES * 16;
    pool_scatter<<<(tPool + 255) / 256, 256, 0, stream>>>(buf0, batch, pool, GIN_N_NODES);
    bn_inplace<<<(npool + 255) / 256, 256, 0, stream>>>(pool, bnf_g, bnf_b, bnf_m, bnf_v,
                                                        npool, GIN_HIDDEN);
    // out[512,32] = pool @ fc.W + fc.b   (2 waves/block: Nout = 32)
    lin_wmma<<<GIN_NGRAPHS / 16, 64, 0, stream>>>(pool, Wslot(7), fc_b,
                                                  nullptr, 0,
                                                  nullptr, nullptr, nullptr, nullptr, 0,
                                                  (float*)d_out, GIN_LATENT);
}
